// TriangleAttention_23476291239962
// MI455X (gfx1250) — matrix operand order repaired and checked by reading, not compiled, not run on hardware
//
#include <hip/hip_runtime.h>
#include <hip/hip_bf16.h>

typedef _Float16 f16;
typedef _Float16 v16h __attribute__((ext_vector_type(16)));
typedef _Float16 h2v  __attribute__((ext_vector_type(2)));
typedef _Float16 h4v  __attribute__((ext_vector_type(4)));
typedef float    v8f  __attribute__((ext_vector_type(8)));

#define NSEQ 256
#define CDIM 128
#define HEADS 4
#define DDIM 32
#define NN (NSEQ * NSEQ)
#define FP16_HUGE_F 32768.0f
#define KEY_SCALE_F 0.17677669529663687f   // 32^-0.5
#define LN_EPS_F 1e-5f

// ---------------- fragment loaders (CDNA5 16-bit WMMA VGPR layouts) ----------
// A (16x32, MxK): lane m = lane&15, hi = lane>>4.
//   VGPR v in 0..3 : K = 2v   + 8*hi
//   VGPR v in 4..7 : K = 2v+8 + 8*hi
// p points at A[m][k0] in row-major storage (K contiguous).
__device__ __forceinline__ v16h load_afrag(const f16* p, int hi) {
    v16h a;
#pragma unroll
    for (int v = 0; v < 8; ++v) {
        int kk = 2 * v + 8 * hi + ((v & 4) ? 8 : 0);
        h2v t = *(const h2v*)(p + kk);
        a[2 * v] = t[0]; a[2 * v + 1] = t[1];
    }
    return a;
}

// B (32x16, KxN): lane n = lane&15, hi = lane>>4. VGPR v: K = 2v + 8*hi (+8 for v>=4).
// p points at Bt[n][k0] where Bt is [N][K] storage (K contiguous).
__device__ __forceinline__ v16h load_bfrag(const f16* p, int hi) {
    v16h b;
#pragma unroll
    for (int v = 0; v < 8; ++v) {
        int kk = 2 * v + 8 * hi + ((v & 4) ? 8 : 0);
        h2v t = *(const h2v*)(p + kk);
        b[2 * v] = t[0]; b[2 * v + 1] = t[1];
    }
    return b;
}

__device__ __forceinline__ float wred_sum(float x) {
#pragma unroll
    for (int o = 16; o > 0; o >>= 1) x += __shfl_xor(x, o, 32);
    return x;
}

// ---------------- K0: weight repack to f16, K-contiguous ----------------
// wt  : [512 col][128 c]   col = sel*128 + h*32 + d ; sel: 0=q(*scale),1=k,2=v,3=gate
// owt : [128 c_out][128 hd]
__global__ void wconv_kernel(const float* __restrict__ qw, const float* __restrict__ kw,
                             const float* __restrict__ vw, const float* __restrict__ gw,
                             const float* __restrict__ ow,
                             f16* __restrict__ wt, f16* __restrict__ owt) {
    int i = blockIdx.x * 256 + threadIdx.x;
    if (i < 512 * 128) {
        int col = i >> 7, c = i & 127;
        int sel = col >> 7, hd = col & 127;
        const float* src = (sel == 0) ? qw : (sel == 1) ? kw : (sel == 2) ? vw : gw;
        float v = src[c * 128 + hd];          // [C,H,D] row-major: c*128 + h*32 + d
        if (sel == 0) v *= KEY_SCALE_F;
        wt[(size_t)col * 128 + c] = (f16)v;
    } else {
        int j = i - 512 * 128;
        if (j < 128 * 128) {
            int c = j >> 7, hd = j & 127;
            owt[(size_t)c * 128 + hd] = (f16)ow[(size_t)hd * 128 + c]; // [H,D,C] -> [C][hd]
        }
    }
}

// ---------------- K1: LayerNorm + f16 act + tri_bias ----------------
// one wave per (q,k) position; 8 waves/block
__global__ void __launch_bounds__(256)
ln_tri_kernel(const float* __restrict__ pact,
              const float* __restrict__ gamma, const float* __restrict__ beta,
              const float* __restrict__ f2w,
              f16* __restrict__ act_h, float* __restrict__ tri) {
    int lane = threadIdx.x & 31, wave = threadIdx.x >> 5;
    int pos = blockIdx.x * 8 + wave;
    const float* xp = pact + (size_t)pos * CDIM + lane * 4;
    float x[4];
    {
        float4 x4 = *(const float4*)xp;
        x[0] = x4.x; x[1] = x4.y; x[2] = x4.z; x[3] = x4.w;
    }
    float s = x[0] + x[1] + x[2] + x[3];
    s = wred_sum(s);
    float mu = s * (1.0f / 128.0f);
    float vs = 0.f;
#pragma unroll
    for (int i = 0; i < 4; ++i) { float d = x[i] - mu; vs += d * d; }
    vs = wred_sum(vs);
    float rstd = rsqrtf(vs * (1.0f / 128.0f) + LN_EPS_F);

    int c0 = lane * 4;
    float y[4];
    h4v yo;
#pragma unroll
    for (int i = 0; i < 4; ++i) {
        y[i] = (x[i] - mu) * rstd * gamma[c0 + i] + beta[c0 + i];
        yo[i] = (f16)y[i];
    }
    *(h4v*)(act_h + (size_t)pos * CDIM + c0) = yo;

    float tb[HEADS] = {0.f, 0.f, 0.f, 0.f};
#pragma unroll
    for (int i = 0; i < 4; ++i) {
#pragma unroll
        for (int h = 0; h < HEADS; ++h) tb[h] += y[i] * f2w[(c0 + i) * HEADS + h];
    }
#pragma unroll
    for (int h = 0; h < HEADS; ++h) {
        tb[h] = wred_sum(tb[h]);
        if (lane == 0) tri[(size_t)h * NN + pos] = tb[h];
    }
}

// ---------------- K2: fused QKVG projection GEMM (per r) ----------------
// [256 x 128] x [128 x 512], register-blocked 16x64 per wave-task:
// one A fragment feeds 4 WMMAs (4 N-tiles). q,k stored [r][n][hd] f16,
// v stored transposed [r][hd][n] f16, gate = sigmoid(.+b) stored [r][n][hd].
// Outer task loop is NOT unrolled: keeps live set ~1 A-frag + 1 B-frag +
// 4 accumulators so nothing spills to scratch.
__global__ void __launch_bounds__(256)
qkvg_kernel(const f16* __restrict__ act_h, const f16* __restrict__ wt,
            const float* __restrict__ gb,
            f16* __restrict__ q_h, f16* __restrict__ k_h,
            f16* __restrict__ vt_h, f16* __restrict__ g_h) {
    int r = blockIdx.x;
    int lane = threadIdx.x & 31, wave = threadIdx.x >> 5;
    int nl = lane & 15, hi = lane >> 4;

#pragma unroll 1
    for (int t = wave; t < 128; t += 8) {          // 16 mt x 8 ngroups
        int mt = t >> 3, ng = t & 7;               // 64-col group; sel uniform per group
        v8f acc[4];
#pragma unroll
        for (int j = 0; j < 4; ++j) acc[j] = (v8f){0.f,0.f,0.f,0.f,0.f,0.f,0.f,0.f};

        const f16* arow = act_h + ((size_t)(r * NSEQ + mt * 16 + nl)) * CDIM;
        const f16* bbase = wt + ((size_t)(ng * 64 + nl)) * CDIM;
#pragma unroll
        for (int kc = 0; kc < 4; ++kc) {
            v16h a = load_afrag(arow + kc * 32, hi);
#pragma unroll
            for (int j = 0; j < 4; ++j) {
                v16h b = load_bfrag(bbase + (size_t)j * 16 * CDIM + kc * 32, hi);
                acc[j] = __builtin_amdgcn_wmma_f32_16x16x32_f16(false, a, false, b,
                                                                (short)0, acc[j], false, false);
            }
        }
        int sel = ng >> 1;                          // 0=q,1=k,2=v,3=gate (uniform)
#pragma unroll
        for (int j = 0; j < 4; ++j) {
            int ncol = ng * 64 + j * 16 + nl;
            int hd = ncol & 127;
#pragma unroll
            for (int v = 0; v < 8; ++v) {
                int row = mt * 16 + v + 8 * hi;
                float val = acc[j][v];
                size_t ridx = ((size_t)(r * NSEQ + row)) * CDIM + hd;
                if (sel == 0) {
                    q_h[ridx] = (f16)val;
                } else if (sel == 1) {
                    k_h[ridx] = (f16)val;
                } else if (sel == 2) {
                    vt_h[((size_t)(r * CDIM + hd)) * NSEQ + row] = (f16)val;
                } else {
                    float g = 1.0f / (1.0f + __expf(-(val + gb[hd])));
                    g_h[ridx] = (f16)g;
                }
            }
        }
    }
}

// ---------------- K3: attention + gating (per r) ----------------
__global__ void __launch_bounds__(256)
attn_kernel(const f16* __restrict__ q_h, const f16* __restrict__ k_h,
            const f16* __restrict__ vt_h, const f16* __restrict__ g_h,
            const float* __restrict__ tri, const float* __restrict__ mask,
            f16* __restrict__ wa_h) {
    __shared__ f16 pbuf[8 * 16 * NSEQ];   // 64 KB: per-wave 16x256 P tile
    int r = blockIdx.x;
    int lane = threadIdx.x & 31, wave = threadIdx.x >> 5;
    int nl = lane & 15, hi = lane >> 4;
    f16* pb = &pbuf[wave * 16 * NSEQ];

#pragma unroll 1
    for (int task = wave; task < 64; task += 8) {
        int h = task >> 4, qt = task & 15;
        int q0 = qt * 16;

        // ---- S = Q K^T  (16 x 256 block, f32 accum in VGPRs) ----
        const f16* qrow = q_h + ((size_t)(r * NSEQ + q0 + nl)) * CDIM + h * DDIM;
        v16h qa = load_afrag(qrow, hi);
        v8f s[16];
#pragma unroll
        for (int kt = 0; kt < 16; ++kt) {
            const f16* kcolp = k_h + ((size_t)(r * NSEQ + kt * 16 + nl)) * CDIM + h * DDIM;
            v16h kb = load_bfrag(kcolp, hi);
            v8f z = {0.f, 0.f, 0.f, 0.f, 0.f, 0.f, 0.f, 0.f};
            s[kt] = __builtin_amdgcn_wmma_f32_16x16x32_f16(false, qa, false, kb,
                                                           (short)0, z, false, false);
        }

        // ---- biases: tri[h][q][k] + FP16_HUGE*(mask[r][k]-1) ----
#pragma unroll
        for (int kt = 0; kt < 16; ++kt) {
            int kcol = kt * 16 + nl;
            float mb = FP16_HUGE_F * (mask[(size_t)r * NSEQ + kcol] - 1.0f);
            const float* tp = tri + (size_t)h * NN + (size_t)q0 * NSEQ + kcol;
#pragma unroll
            for (int v = 0; v < 8; ++v) {
                int m = v + 8 * hi;
                s[kt][v] += tp[(size_t)m * NSEQ] + mb;
            }
        }

        // ---- softmax over k (rows live in 16-lane halves) ----
        float mx[8], sm[8];
#pragma unroll
        for (int v = 0; v < 8; ++v) {
            float m = -3.4e38f;
#pragma unroll
            for (int kt = 0; kt < 16; ++kt) m = fmaxf(m, s[kt][v]);
#pragma unroll
            for (int o = 8; o > 0; o >>= 1) m = fmaxf(m, __shfl_xor(m, o, 32));
            mx[v] = m;
        }
#pragma unroll
        for (int v = 0; v < 8; ++v) {
            float acc = 0.f;
#pragma unroll
            for (int kt = 0; kt < 16; ++kt) {
                s[kt][v] = __expf(s[kt][v] - mx[v]);
                acc += s[kt][v];
            }
#pragma unroll
            for (int o = 8; o > 0; o >>= 1) acc += __shfl_xor(acc, o, 32);
            sm[v] = 1.0f / acc;
        }

        // ---- P -> LDS (D-layout -> row-major f16) ----
#pragma unroll
        for (int kt = 0; kt < 16; ++kt) {
#pragma unroll
            for (int v = 0; v < 8; ++v) {
                pb[(size_t)(v + 8 * hi) * NSEQ + kt * 16 + nl] = (f16)(s[kt][v] * sm[v]);
            }
        }
        asm volatile("s_wait_dscnt 0" ::: "memory");

        // ---- O = P V  (16 x 32) ----
        v8f o0 = {0.f, 0.f, 0.f, 0.f, 0.f, 0.f, 0.f, 0.f};
        v8f o1 = {0.f, 0.f, 0.f, 0.f, 0.f, 0.f, 0.f, 0.f};
#pragma unroll
        for (int kc = 0; kc < 8; ++kc) {
            v16h pa = load_afrag(pb + (size_t)nl * NSEQ + kc * 32, hi);
            const f16* v0p = vt_h + ((size_t)(r * CDIM + h * DDIM + nl)) * NSEQ + kc * 32;
            const f16* v1p = vt_h + ((size_t)(r * CDIM + h * DDIM + 16 + nl)) * NSEQ + kc * 32;
            v16h vb0 = load_bfrag(v0p, hi);
            v16h vb1 = load_bfrag(v1p, hi);
            o0 = __builtin_amdgcn_wmma_f32_16x16x32_f16(false, pa, false, vb0,
                                                        (short)0, o0, false, false);
            o1 = __builtin_amdgcn_wmma_f32_16x16x32_f16(false, pa, false, vb1,
                                                        (short)0, o1, false, false);
        }

        // ---- gate and store wa ----
#pragma unroll
        for (int v = 0; v < 8; ++v) {
            int qg = q0 + v + 8 * hi;
            int hd0 = h * DDIM + nl, hd1 = h * DDIM + 16 + nl;
            size_t base = ((size_t)(r * NSEQ + qg)) * CDIM;
            float g0 = (float)g_h[base + hd0];
            float g1 = (float)g_h[base + hd1];
            wa_h[base + hd0] = (f16)(o0[v] * g0);
            wa_h[base + hd1] = (f16)(o1[v] * g1);
        }
    }
}

// ---------------- K4: output projection GEMM (per r) ----------------
// register-blocked 16x64 per wave-task: one A fragment feeds 4 WMMAs.
__global__ void __launch_bounds__(256)
out_kernel(const f16* __restrict__ wa_h, const f16* __restrict__ owt,
           const float* __restrict__ ob, float* __restrict__ out) {
    int r = blockIdx.x;
    int lane = threadIdx.x & 31, wave = threadIdx.x >> 5;
    int nl = lane & 15, hi = lane >> 4;

#pragma unroll 1
    for (int t = wave; t < 32; t += 8) {           // 16 mt x 2 ngroups
        int mt = t >> 1, ng = t & 1;
        v8f acc[4];
#pragma unroll
        for (int j = 0; j < 4; ++j) acc[j] = (v8f){0.f,0.f,0.f,0.f,0.f,0.f,0.f,0.f};

        const f16* arow = wa_h + ((size_t)(r * NSEQ + mt * 16 + nl)) * CDIM;
        const f16* bbase = owt + ((size_t)(ng * 64 + nl)) * CDIM;
#pragma unroll
        for (int kc = 0; kc < 4; ++kc) {
            v16h a = load_afrag(arow + kc * 32, hi);
#pragma unroll
            for (int j = 0; j < 4; ++j) {
                v16h b = load_bfrag(bbase + (size_t)j * 16 * CDIM + kc * 32, hi);
                acc[j] = __builtin_amdgcn_wmma_f32_16x16x32_f16(false, a, false, b,
                                                                (short)0, acc[j], false, false);
            }
        }
#pragma unroll
        for (int j = 0; j < 4; ++j) {
            int ncol = ng * 64 + j * 16 + nl;
            float bias = ob[ncol];
#pragma unroll
            for (int v = 0; v < 8; ++v) {
                int row = mt * 16 + v + 8 * hi;
                out[((size_t)(r * NSEQ + row)) * CDIM + ncol] = acc[j][v] + bias;
            }
        }
    }
}

// ---------------- launch ----------------
extern "C" void kernel_launch(void* const* d_in, const int* in_sizes, int n_in,
                              void* d_out, int out_size, void* d_ws, size_t ws_size,
                              hipStream_t stream) {
    (void)in_sizes; (void)n_in; (void)out_size; (void)ws_size;
    const float* pact  = (const float*)d_in[0];
    const float* mask  = (const float*)d_in[1];
    const float* gamma = (const float*)d_in[2];
    const float* beta  = (const float*)d_in[3];
    const float* f2w   = (const float*)d_in[4];
    const float* qw    = (const float*)d_in[5];
    const float* kw    = (const float*)d_in[6];
    const float* vw    = (const float*)d_in[7];
    const float* gw    = (const float*)d_in[8];
    const float* gb    = (const float*)d_in[9];
    const float* ow    = (const float*)d_in[10];
    const float* ob    = (const float*)d_in[11];

    char* ws = (char*)d_ws;
    size_t o = 0;
    const size_t actBytes = (size_t)NN * CDIM * sizeof(f16);   // 16 MB each
    f16* act_h = (f16*)(ws + o); o += actBytes;
    f16* q_h   = (f16*)(ws + o); o += actBytes;
    f16* k_h   = (f16*)(ws + o); o += actBytes;
    f16* vt_h  = (f16*)(ws + o); o += actBytes;
    f16* g_h   = (f16*)(ws + o); o += actBytes;
    f16* wa_h  = (f16*)(ws + o); o += actBytes;
    float* tri = (float*)(ws + o); o += (size_t)HEADS * NN * sizeof(float);
    f16* wt    = (f16*)(ws + o); o += (size_t)512 * 128 * sizeof(f16);
    f16* owt   = (f16*)(ws + o); o += (size_t)128 * 128 * sizeof(f16);

    wconv_kernel<<<320, 256, 0, stream>>>(qw, kw, vw, gw, ow, wt, owt);
    ln_tri_kernel<<<NN / 8, 256, 0, stream>>>(pact, gamma, beta, f2w, act_h, tri);
    qkvg_kernel<<<NSEQ, 256, 0, stream>>>(act_h, wt, gb, q_h, k_h, vt_h, g_h);
    attn_kernel<<<NSEQ, 256, 0, stream>>>(q_h, k_h, vt_h, g_h, tri, mask, wa_h);
    out_kernel<<<NSEQ, 256, 0, stream>>>(wa_h, owt, ob, (float*)d_out);
}
